// EvaBlock_40656160424186
// MI455X (gfx1250) — compile-verified, matrix operand order checked by reading, not run on hardware
//
#include <hip/hip_runtime.h>
#include <math.h>

// Problem constants (from reference): B=16, N=1024, C=768, NH=12, HD=64
#define BDIM 16
#define NSEQ 1024
#define CDIM 768
#define NHEAD 12
#define HD 64
#define MROWS (BDIM * NSEQ)   // 16384
#define C3 (3 * CDIM)         // 2304

typedef __attribute__((ext_vector_type(16))) __bf16 v16bf;
typedef __attribute__((ext_vector_type(8)))  float  v8f;
typedef __attribute__((ext_vector_type(4)))  unsigned int u32x4;
typedef __attribute__((ext_vector_type(8)))  int i32x8;
typedef __attribute__((ext_vector_type(4)))  int i32x4;

union FragU { v16bf v; unsigned int u[8]; };

__device__ __forceinline__ unsigned short f2bf(float f) {
  unsigned int u = __float_as_uint(f);
  return (unsigned short)((u + 0x7FFFu + ((u >> 16) & 1u)) >> 16);  // RNE
}
__device__ __forceinline__ float bf2f(unsigned short h) {
  return __uint_as_float(((unsigned int)h) << 16);
}

// Load a 16x32 bf16 fragment from LDS (rows contiguous along K).
// Per CDNA5 ISA 16-bit A layout: lane L holds row (L&15); VGPR p holds
// K = (L>>4)*8 + (p&3)*2 [+16 for p>=4], pairs packed per dword.
// B fragments use the same shape with the LDS tile stored [n][k].
__device__ __forceinline__ v16bf load_frag(const unsigned short* row0, int rowStride, int lane) {
  FragU f;
  const unsigned short* row = row0 + (size_t)(lane & 15) * rowStride;
  const int hi = (lane >> 4) * 8;
#pragma unroll
  for (int p = 0; p < 8; ++p) {
    const int kk = hi + ((p & 3) << 1) + ((p & 4) << 2);  // +16 when p>=4
    f.u[p] = *reinterpret_cast<const unsigned int*>(row + kk);
  }
  return f.v;
}

__device__ __forceinline__ v8f wmma_bf(v16bf a, v16bf b, v8f c) {
  return __builtin_amdgcn_wmma_f32_16x16x32_bf16(false, a, false, b, (short)0, c, false, false);
}

__device__ __forceinline__ float gelu_exact(float x) {
  return 0.5f * x * (1.0f + erff(x * 0.70710678118654752f));
}

__device__ __forceinline__ float wave_sum(float s) {
#pragma unroll
  for (int off = 16; off > 0; off >>= 1) s += __shfl_xor(s, off, 32);
  return s;
}

// ---------------------------------------------------------------------------
// CDNA5 async global->LDS copy (16B per lane), tracked by ASYNCcnt.
// ---------------------------------------------------------------------------
__device__ __forceinline__ void async_copy_b128(const unsigned short* gptr,
                                                const unsigned short* lptr) {
  unsigned int l = (unsigned int)(uintptr_t)lptr;         // LDS byte offset (low 32 bits)
  unsigned long long g = (unsigned long long)(uintptr_t)gptr;
  asm volatile("global_load_async_to_lds_b128 %0, %1, off" :: "v"(l), "v"(g) : "memory");
}
__device__ __forceinline__ void wait_async() {
  asm volatile("s_wait_asynccnt 0" ::: "memory");
}

// ---------------------------------------------------------------------------
// Tensor Data Mover: 2D tile (tile0 x tile1 elements of 2 bytes) from global
// (row stride stride0 elements) into LDS at lds_off, rows packed contiguously.
// D# per CDNA5 ISA 8.3/8.4: group0 = {count/lds/global/type}, group1 = dims.
// This toolchain exposes the 6-arg builtin: (g0, g1, g2, g3, g4, cpol).
// ---------------------------------------------------------------------------
__device__ __forceinline__ void tdm_load_2d_bf16(const void* gaddr, unsigned int lds_off,
                                                 unsigned int tile0, unsigned int tile1,
                                                 unsigned long long stride0) {
  unsigned long long ga = (unsigned long long)(uintptr_t)gaddr;
  const unsigned int td0 = 0x40000000u;  // generous tensor extents (no OOB clipping)
  const unsigned int td1 = 0x40000000u;
  u32x4 g0;
  g0.x = 1u;                                               // count=1 (user descriptor)
  g0.y = lds_off;                                          // lds_addr
  g0.z = (unsigned int)(ga & 0xFFFFFFFFu);                 // global_addr[31:0]
  g0.w = (unsigned int)((ga >> 32) & 0x01FFFFFFu) | (2u << 30);  // addr[56:32] | type=2
  i32x8 g1;
  g1[0] = (int)(1u << 16);                                 // data_size=1 (2 bytes)
  g1[1] = (int)((td0 & 0xFFFFu) << 16);                    // tensor_dim0[15:0]
  g1[2] = (int)(((td0 >> 16) & 0xFFFFu) | ((td1 & 0xFFFFu) << 16));
  g1[3] = (int)(((td1 >> 16) & 0xFFFFu) | ((tile0 & 0xFFFFu) << 16));  // tile_dim0
  g1[4] = (int)(tile1 & 0xFFFFu);                          // tile_dim1 (tile_dim2 = 0)
  g1[5] = (int)(unsigned int)(stride0 & 0xFFFFFFFFu);      // tensor_dim0_stride[31:0]
  g1[6] = (int)(unsigned int)((stride0 >> 32) & 0xFFFFu);  // stride0[47:32], dim1_stride=0
  g1[7] = 0;
  i32x4 z4 = {0, 0, 0, 0};
  i32x8 z8 = {0, 0, 0, 0, 0, 0, 0, 0};
  __builtin_amdgcn_tensor_load_to_lds(g0, g1, z4, z4, z8, 0);
}

// ---------------------------------------------------------------------------
// Elementwise helpers
// ---------------------------------------------------------------------------
__global__ void cvt_bf16_kernel(const float* __restrict__ in, unsigned short* __restrict__ o, int n) {
  int i = blockIdx.x * blockDim.x + threadIdx.x;
  if (i < n) o[i] = f2bf(in[i]);
}

__global__ void zero_kernel(float* __restrict__ p, int n) {
  int i = blockIdx.x * blockDim.x + threadIdx.x;
  if (i < n) p[i] = 0.0f;
}

__global__ void qkv_bias_kernel(const float* __restrict__ qb, const float* __restrict__ vb,
                                float* __restrict__ o) {
  int i = blockIdx.x * blockDim.x + threadIdx.x;
  if (i >= C3) return;
  float v = 0.0f;
  if (i < CDIM) v = qb[i];
  else if (i >= 2 * CDIM) v = vb[i - 2 * CDIM];
  o[i] = v;
}

// LayerNorm over C=768, one row per block of 256 threads; bf16 output.
__global__ __launch_bounds__(256) void ln_kernel(const float* __restrict__ x,
                                                 const float* __restrict__ g,
                                                 const float* __restrict__ bb,
                                                 unsigned short* __restrict__ outbf) {
  __shared__ float sred[256];
  const int row = blockIdx.x, tid = threadIdx.x;
  const float* xr = x + (size_t)row * CDIM;
  const float v0 = xr[tid], v1 = xr[tid + 256], v2 = xr[tid + 512];
  sred[tid] = v0 + v1 + v2;
  __syncthreads();
  for (int o = 128; o > 0; o >>= 1) { if (tid < o) sred[tid] += sred[tid + o]; __syncthreads(); }
  const float mu = sred[0] * (1.0f / CDIM);
  __syncthreads();
  const float d0 = v0 - mu, d1 = v1 - mu, d2 = v2 - mu;
  sred[tid] = d0 * d0 + d1 * d1 + d2 * d2;
  __syncthreads();
  for (int o = 128; o > 0; o >>= 1) { if (tid < o) sred[tid] += sred[tid + o]; __syncthreads(); }
  const float rstd = rsqrtf(sred[0] * (1.0f / CDIM) + 1e-6f);
  unsigned short* orow = outbf + (size_t)row * CDIM;
  orow[tid]       = f2bf(g[tid]       * d0 * rstd + bb[tid]);
  orow[tid + 256] = f2bf(g[tid + 256] * d1 * rstd + bb[tid + 256]);
  orow[tid + 512] = f2bf(g[tid + 512] * d2 * rstd + bb[tid + 512]);
}

// eta[b,h,n] = sigmoid(h . lr_w[h] + lr_b[h]) / HD   (one wave per (b,h,n))
__global__ __launch_bounds__(256) void lr_kernel(const unsigned short* __restrict__ hbf,
                                                 const float* __restrict__ lrw,
                                                 const float* __restrict__ lrb,
                                                 float* __restrict__ eta) {
  const int tid = threadIdx.x, lane = tid & 31, wid = tid >> 5;
  const size_t idx = (size_t)blockIdx.x * 8 + wid;  // over (b, h, n)
  const int b = (int)(idx / (NHEAD * NSEQ));
  const int h = (int)((idx / NSEQ) % NHEAD);
  const int n = (int)(idx % NSEQ);
  const unsigned short* hr = hbf + (size_t)(b * NSEQ + n) * CDIM;
  const float* wr = lrw + (size_t)h * CDIM;
  float s = 0.0f;
#pragma unroll
  for (int i = 0; i < CDIM / 32; ++i) {
    const int c = lane + i * 32;
    s += bf2f(hr[c]) * wr[c];
  }
  s = wave_sum(s);
  if (lane == 0) {
    const float sg = 1.0f / (1.0f + __expf(-(s + lrb[h])));
    eta[idx] = sg * (1.0f / HD);
  }
}

// ---------------------------------------------------------------------------
// Generic big GEMM: out = act(A[M,K](bf16) * W[N,K]^T(bf16) + bias [+residual])
// Block tile 128x128, 8 waves (4x2), wave tile 32x64 = 2x4 WMMA frags.
// Tile staging via async global->LDS (ASYNCcnt).
// ---------------------------------------------------------------------------
__global__ __launch_bounds__(256) void gemm_bf16_kernel(
    const unsigned short* __restrict__ A, int lda,
    const unsigned short* __restrict__ W,
    const float* __restrict__ bias,
    const float* __restrict__ residual,
    float* __restrict__ outF,
    unsigned short* __restrict__ outB,
    int M, int Nout, int K, int act) {
  __shared__ __align__(16) unsigned short As[128][32];
  __shared__ __align__(16) unsigned short Bs[128][32];
  const int tid = threadIdx.x, lane = tid & 31, wid = tid >> 5;
  const int bm = blockIdx.x, bn = blockIdx.y;
  const int wm = wid >> 1, wn = wid & 1;
  v8f acc[2][4] = {};
  const int r = tid >> 1, hf = tid & 1;
  const unsigned short* aBase = A + (size_t)(bm * 128 + r) * lda + hf * 16;
  const unsigned short* wBase = W + (size_t)(bn * 128 + r) * K + hf * 16;
  unsigned short* aL = &As[r][hf * 16];
  unsigned short* bL = &Bs[r][hf * 16];

  for (int k0 = 0; k0 < K; k0 += 32) {
    async_copy_b128(aBase + k0,     aL);
    async_copy_b128(aBase + k0 + 8, aL + 8);
    async_copy_b128(wBase + k0,     bL);
    async_copy_b128(wBase + k0 + 8, bL + 8);
    if (k0 + 32 < K) __builtin_prefetch(aBase + k0 + 32, 0, 1);
    wait_async();
    __syncthreads();
    v16bf a0 = load_frag(&As[wm * 32][0], 32, lane);
    v16bf a1 = load_frag(&As[wm * 32 + 16][0], 32, lane);
#pragma unroll
    for (int j = 0; j < 4; ++j) {
      v16bf b = load_frag(&Bs[wn * 64 + j * 16][0], 32, lane);
      acc[0][j] = wmma_bf(a0, b, acc[0][j]);
      acc[1][j] = wmma_bf(a1, b, acc[1][j]);
    }
    __syncthreads();
  }

  const int r0 = (lane >> 4) * 8, cn = lane & 15;
#pragma unroll
  for (int fi = 0; fi < 2; ++fi) {
    const int gmB = bm * 128 + wm * 32 + fi * 16 + r0;
#pragma unroll
    for (int fj = 0; fj < 4; ++fj) {
      const int gn = bn * 128 + wn * 64 + fj * 16 + cn;
      const float bv = bias ? bias[gn] : 0.0f;
#pragma unroll
      for (int i = 0; i < 8; ++i) {
        const size_t gm = (size_t)(gmB + i);
        float v = acc[fi][fj][i] + bv;
        if (residual) v += residual[gm * Nout + gn];
        if (act) v = gelu_exact(v);
        if (outF) outF[gm * Nout + gn] = v;
        if (outB) outB[gm * Nout + gn] = f2bf(v);
      }
    }
  }
}

// ---------------------------------------------------------------------------
// Per-head GEMM: out[bh, n, e] = A[b*N+n, colBase + h*HD + d] * Bsm[d][e] + bias[e]
// A tile (128x64, row stride lda) staged by the Tensor Data Mover (TENSORcnt).
// Block: 128 rows x 64 cols; 8 waves, each wave 16 rows x 64 cols (8 WMMAs).
// ---------------------------------------------------------------------------
__global__ __launch_bounds__(256) void head_gemm_kernel(
    const unsigned short* __restrict__ Abf, int lda, int colBase,
    const unsigned short* __restrict__ Bsm, int bStrB, int bStrH,
    const float* __restrict__ bias, int biasStrB, int biasStrH,
    float* __restrict__ out) {
  __shared__ __align__(16) unsigned short As[128][64];
  __shared__ __align__(16) unsigned short Bs[64][64];  // [e][d]  (transposed)
  const int tid = threadIdx.x, lane = tid & 31, wid = tid >> 5;
  const int bh = blockIdx.x, b = bh / NHEAD, h = bh % NHEAD;
  const int n0 = blockIdx.y * 128;
  // TDM: one 2D tile load (64 x 128 rows) issued by wave 0
  const unsigned short* gA = Abf + (size_t)(b * NSEQ + n0) * lda + colBase + h * HD;
  if (tid < 32) {
    tdm_load_2d_bf16(gA, (unsigned int)(uintptr_t)&As[0][0], HD, 128,
                     (unsigned long long)lda);
    __builtin_amdgcn_s_wait_tensorcnt(0);
  }
  // Stage Bsm transposed: Bs[e][d] = Bsm[d][e]
  const unsigned short* Bp = Bsm + (size_t)b * bStrB + (size_t)h * bStrH;  // [d][e]
  for (int i = tid; i < HD * HD; i += 256) Bs[i & 63][i >> 6] = Bp[i];
  __syncthreads();

  v8f acc[4] = {};
#pragma unroll
  for (int ks = 0; ks < HD; ks += 32) {
    v16bf a = load_frag(&As[wid * 16][ks], 64, lane);
#pragma unroll
    for (int j = 0; j < 4; ++j) {
      v16bf bf = load_frag(&Bs[j * 16][ks], 64, lane);
      acc[j] = wmma_bf(a, bf, acc[j]);
    }
  }
  const int r0 = (lane >> 4) * 8, cn = lane & 15;
  const float* bp = bias + (size_t)b * biasStrB + (size_t)h * biasStrH;
  float* orow = out + ((size_t)bh * NSEQ + n0 + wid * 16) * HD;
#pragma unroll
  for (int j = 0; j < 4; ++j) {
    const int e = j * 16 + cn;
    const float bv = bp[e];
#pragma unroll
    for (int i = 0; i < 8; ++i) {
      orow[(size_t)(r0 + i) * HD + e] = acc[j][i] + bv;
    }
  }
}

// ---------------------------------------------------------------------------
// Fused LN-L2-backward per row (b,h,n): grad_z from Z1, tgt = v-k; eg = eta*grad.
// One wave per row (2 cols/lane); LDS + global f32 atomics for column sums.
// ---------------------------------------------------------------------------
__global__ __launch_bounds__(256) void gradz_kernel(
    const float* __restrict__ Z1,
    const unsigned short* __restrict__ qkvbf,
    const float* __restrict__ eta,
    const float* __restrict__ ttt_w, const float* __restrict__ ttt_b,
    unsigned short* __restrict__ egbf,
    float* __restrict__ egsum) {
  __shared__ float colsum[HD];
  const int tid = threadIdx.x, lane = tid & 31, wid = tid >> 5;
  if (tid < HD) colsum[tid] = 0.0f;
  __syncthreads();
  const size_t ridx = (size_t)blockIdx.x * 8 + wid;  // (b,h,n), 1024 % 8 == 0
  const int b = (int)(ridx / (NHEAD * NSEQ));
  const int h = (int)((ridx / NSEQ) % NHEAD);
  const int n = (int)(ridx % NSEQ);
  const int c0 = lane, c1 = lane + 32;
  const float* zr = Z1 + ridx * HD;
  float z0 = zr[c0], z1 = zr[c1];
  const size_t qb = (size_t)(b * NSEQ + n) * C3 + h * HD;
  float k0 = bf2f(qkvbf[qb + CDIM + c0]),     k1 = bf2f(qkvbf[qb + CDIM + c1]);
  float v0 = bf2f(qkvbf[qb + 2 * CDIM + c0]), v1 = bf2f(qkvbf[qb + 2 * CDIM + c1]);
  float t0 = v0 - k0, t1 = v1 - k1;
  float gw0 = ttt_w[h * HD + c0], gw1 = ttt_w[h * HD + c1];
  float gb0 = ttt_b[h * HD + c0], gb1 = ttt_b[h * HD + c1];
  float mu = wave_sum(z0 + z1) * (1.0f / HD);
  float d0 = z0 - mu, d1 = z1 - mu;
  float var = wave_sum(d0 * d0 + d1 * d1) * (1.0f / HD);
  float stdv = sqrtf(var + 1e-6f);
  float rstd = 1.0f / stdv;
  float xh0 = d0 * rstd, xh1 = d1 * rstd;
  float gho0 = (gw0 * xh0 + gb0 - t0) * gw0;
  float gho1 = (gw1 * xh1 + gb1 - t1) * gw1;
  float s1 = wave_sum(gho0 + gho1);
  float s2 = wave_sum(gho0 * xh0 + gho1 * xh1);
  float den = 1.0f / ((float)HD * stdv);
  float g0 = ((float)HD * gho0 - s1 - xh0 * s2) * den;
  float g1 = ((float)HD * gho1 - s1 - xh1 * s2) * den;
  float et = eta[ridx];
  float e0 = et * g0, e1 = et * g1;
  unsigned short* er = egbf + ridx * HD;
  er[c0] = f2bf(e0); er[c1] = f2bf(e1);
  atomicAdd(&colsum[c0], e0);
  atomicAdd(&colsum[c1], e1);
  __syncthreads();
  if (tid < HD) atomicAdd(&egsum[(size_t)(b * NHEAD + h) * HD + tid], colsum[tid]);
}

// ---------------------------------------------------------------------------
// Per-(b,h): G = K^T(64x1024) * EG(1024x64) via WMMA (both staged transposed in
// LDS). W1_bar = W1 - G (bf16 out); b1_bar = b1 - egsum.
// ---------------------------------------------------------------------------
__global__ __launch_bounds__(256) void w1bar_kernel(
    const unsigned short* __restrict__ qkvbf,
    const unsigned short* __restrict__ egbf,
    const float* __restrict__ W1,
    const float* __restrict__ b1,
    const float* __restrict__ egsum,
    float* __restrict__ b1bar,
    unsigned short* __restrict__ w1barbf) {
  __shared__ __align__(16) unsigned short Kt[64][32];  // [d][n]
  __shared__ __align__(16) unsigned short Et[64][32];  // [e][n]
  const int tid = threadIdx.x, lane = tid & 31, wid = tid >> 5;
  const int bh = blockIdx.x, b = bh / NHEAD, h = bh % NHEAD;
  if (tid < HD)
    b1bar[(size_t)bh * HD + tid] = b1[h * HD + tid] - egsum[(size_t)bh * HD + tid];
  const int fi = wid >> 1, fj0 = (wid & 1) * 2;
  v8f acc[2] = {};
  for (int n0 = 0; n0 < NSEQ; n0 += 32) {
    for (int i = tid; i < 2048; i += 256) {
      const int n = i >> 6, d = i & 63;
      Kt[d][n] = qkvbf[(size_t)(b * NSEQ + n0 + n) * C3 + CDIM + h * HD + d];
      Et[d][n] = egbf[((size_t)bh * NSEQ + n0 + n) * HD + d];
    }
    __syncthreads();
    v16bf a  = load_frag(&Kt[fi * 16][0], 32, lane);
    v16bf e0 = load_frag(&Et[fj0 * 16][0], 32, lane);
    v16bf e1 = load_frag(&Et[(fj0 + 1) * 16][0], 32, lane);
    acc[0] = wmma_bf(a, e0, acc[0]);
    acc[1] = wmma_bf(a, e1, acc[1]);
    __syncthreads();
  }
  const int r0 = (lane >> 4) * 8, cn = lane & 15;
#pragma unroll
  for (int jj = 0; jj < 2; ++jj) {
    const int e = (fj0 + jj) * 16 + cn;
#pragma unroll
    for (int i = 0; i < 8; ++i) {
      const int d = fi * 16 + r0 + i;
      const float w = W1[(size_t)h * HD * HD + d * HD + e] - acc[jj][i];
      w1barbf[(size_t)bh * HD * HD + d * HD + e] = f2bf(w);
    }
  }
}

// attn[b,n,h*HD+d] = q + LN(Z; ttt_w[h], ttt_b[h]); one wave per (b,h,n).
__global__ __launch_bounds__(256) void attn_ln_kernel(
    const float* __restrict__ Z,
    const unsigned short* __restrict__ qkvbf,
    const float* __restrict__ ttt_w, const float* __restrict__ ttt_b,
    unsigned short* __restrict__ attnbf) {
  const int tid = threadIdx.x, lane = tid & 31, wid = tid >> 5;
  const size_t ridx = (size_t)blockIdx.x * 8 + wid;
  const int b = (int)(ridx / (NHEAD * NSEQ));
  const int h = (int)((ridx / NSEQ) % NHEAD);
  const int n = (int)(ridx % NSEQ);
  const int c0 = lane, c1 = lane + 32;
  const float* zr = Z + ridx * HD;
  float z0 = zr[c0], z1 = zr[c1];
  float mu = wave_sum(z0 + z1) * (1.0f / HD);
  float d0 = z0 - mu, d1 = z1 - mu;
  float var = wave_sum(d0 * d0 + d1 * d1) * (1.0f / HD);
  float rstd = rsqrtf(var + 1e-6f);
  const size_t qb = (size_t)(b * NSEQ + n) * C3 + h * HD;
  float q0 = bf2f(qkvbf[qb + c0]), q1 = bf2f(qkvbf[qb + c1]);
  float a0 = q0 + ttt_w[h * HD + c0] * d0 * rstd + ttt_b[h * HD + c0];
  float a1 = q1 + ttt_w[h * HD + c1] * d1 * rstd + ttt_b[h * HD + c1];
  unsigned short* o = attnbf + (size_t)(b * NSEQ + n) * CDIM + h * HD;
  o[c0] = f2bf(a0); o[c1] = f2bf(a1);
}

// ---------------------------------------------------------------------------
extern "C" void kernel_launch(void* const* d_in, const int* in_sizes, int n_in,
                              void* d_out, int out_size, void* d_ws, size_t ws_size,
                              hipStream_t stream) {
  (void)in_sizes; (void)n_in; (void)out_size; (void)ws_size;
  const float* x      = (const float*)d_in[0];
  const float* qkv_w  = (const float*)d_in[1];
  const float* q_bias = (const float*)d_in[2];
  const float* v_bias = (const float*)d_in[3];
  const float* proj_w = (const float*)d_in[4];
  const float* proj_b = (const float*)d_in[5];
  const float* lr_w   = (const float*)d_in[6];
  const float* lr_b   = (const float*)d_in[7];
  const float* W1     = (const float*)d_in[8];
  const float* b1     = (const float*)d_in[9];
  const float* ttt_w  = (const float*)d_in[10];
  const float* ttt_b  = (const float*)d_in[11];
  const float* n1w    = (const float*)d_in[12];
  const float* n1b    = (const float*)d_in[13];
  const float* n2w    = (const float*)d_in[14];
  const float* n2b    = (const float*)d_in[15];
  const float* fc1_w  = (const float*)d_in[16];
  const float* fc1_b  = (const float*)d_in[17];
  const float* fc2_w  = (const float*)d_in[18];
  const float* fc2_b  = (const float*)d_in[19];
  float* out = (float*)d_out;

  char* ws = (char*)d_ws;
  size_t off = 0;
  auto alloc = [&](size_t bytes) -> void* {
    void* p = ws + off;
    off += (bytes + 255) & ~(size_t)255;
    return p;
  };
  unsigned short* hbf      = (unsigned short*)alloc((size_t)MROWS * CDIM * 2);
  unsigned short* qkvw_bf  = (unsigned short*)alloc((size_t)C3 * CDIM * 2);
  unsigned short* projw_bf = (unsigned short*)alloc((size_t)CDIM * CDIM * 2);
  unsigned short* fc1w_bf  = (unsigned short*)alloc((size_t)4 * CDIM * CDIM * 2);
  unsigned short* fc2w_bf  = (unsigned short*)alloc((size_t)4 * CDIM * CDIM * 2);
  unsigned short* W1_bf    = (unsigned short*)alloc((size_t)NHEAD * HD * HD * 2);
  float*          qkvbias  = (float*)alloc((size_t)C3 * 4);
  float*          eta      = (float*)alloc((size_t)BDIM * NHEAD * NSEQ * 4);
  unsigned short* qkv_bf   = (unsigned short*)alloc((size_t)MROWS * C3 * 2);
  float*          zbuf     = (float*)alloc((size_t)BDIM * NHEAD * NSEQ * HD * 4);
  unsigned short* eg_bf    = (unsigned short*)alloc((size_t)BDIM * NHEAD * NSEQ * HD * 2);
  float*          egsum    = (float*)alloc((size_t)BDIM * NHEAD * HD * 4);
  unsigned short* w1bar_bf = (unsigned short*)alloc((size_t)BDIM * NHEAD * HD * HD * 2);
  float*          b1bar    = (float*)alloc((size_t)BDIM * NHEAD * HD * 4);
  unsigned short* attn_bf  = (unsigned short*)alloc((size_t)MROWS * CDIM * 2);
  float*          x2       = (float*)alloc((size_t)MROWS * CDIM * 4);
  unsigned short* h2bf     = (unsigned short*)alloc((size_t)MROWS * CDIM * 2);
  unsigned short* fc1a_bf  = (unsigned short*)alloc((size_t)MROWS * 4 * CDIM * 2);

  const int NBH  = BDIM * NHEAD;                // 192
  const int NROW = BDIM * NHEAD * NSEQ;         // 196608

  // 1) Weight conversions + bias assembly
  cvt_bf16_kernel<<<(C3 * CDIM + 255) / 256, 256, 0, stream>>>(qkv_w, qkvw_bf, C3 * CDIM);
  cvt_bf16_kernel<<<(CDIM * CDIM + 255) / 256, 256, 0, stream>>>(proj_w, projw_bf, CDIM * CDIM);
  cvt_bf16_kernel<<<(4 * CDIM * CDIM + 255) / 256, 256, 0, stream>>>(fc1_w, fc1w_bf, 4 * CDIM * CDIM);
  cvt_bf16_kernel<<<(4 * CDIM * CDIM + 255) / 256, 256, 0, stream>>>(fc2_w, fc2w_bf, 4 * CDIM * CDIM);
  cvt_bf16_kernel<<<(NHEAD * HD * HD + 255) / 256, 256, 0, stream>>>(W1, W1_bf, NHEAD * HD * HD);
  qkv_bias_kernel<<<(C3 + 255) / 256, 256, 0, stream>>>(q_bias, v_bias, qkvbias);

  // 2) LN1 + ttt learning rates
  ln_kernel<<<MROWS, 256, 0, stream>>>(x, n1w, n1b, hbf);
  lr_kernel<<<NROW / 8, 256, 0, stream>>>(hbf, lr_w, lr_b, eta);

  // 3) QKV GEMM (16384 x 2304 x 768), bf16 out
  gemm_bf16_kernel<<<dim3(MROWS / 128, C3 / 128), 256, 0, stream>>>(
      hbf, CDIM, qkvw_bf, qkvbias, nullptr, nullptr, qkv_bf, MROWS, C3, CDIM, 0);

  // 4) Z1 = K @ W1 + b1 per head
  head_gemm_kernel<<<dim3(NBH, NSEQ / 128), 256, 0, stream>>>(
      qkv_bf, C3, CDIM, W1_bf, 0, HD * HD, b1, 0, HD, zbuf);

  // 5) grad_z / eg + column sums
  zero_kernel<<<(NBH * HD + 255) / 256, 256, 0, stream>>>(egsum, NBH * HD);
  gradz_kernel<<<NROW / 8, 256, 0, stream>>>(zbuf, qkv_bf, eta, ttt_w, ttt_b, eg_bf, egsum);

  // 6) W1_bar = W1 - K^T @ EG ; b1_bar = b1 - sum(eg)
  w1bar_kernel<<<NBH, 256, 0, stream>>>(qkv_bf, eg_bf, W1, b1, egsum, b1bar, w1bar_bf);

  // 7) Z = Q @ W1_bar + b1_bar (reuse zbuf), then attn = q + LN(Z)
  head_gemm_kernel<<<dim3(NBH, NSEQ / 128), 256, 0, stream>>>(
      qkv_bf, C3, 0, w1bar_bf, NHEAD * HD * HD, HD * HD, b1bar, NHEAD * HD, HD, zbuf);
  attn_ln_kernel<<<NROW / 8, 256, 0, stream>>>(zbuf, qkv_bf, ttt_w, ttt_b, attn_bf);

  // 8) x2 = x + attn @ proj_w^T + proj_b
  gemm_bf16_kernel<<<dim3(MROWS / 128, CDIM / 128), 256, 0, stream>>>(
      attn_bf, CDIM, projw_bf, proj_b, x, x2, nullptr, MROWS, CDIM, CDIM, 0);

  // 9) MLP: h2 = LN(x2); fc1(+gelu); out = x2 + fc2
  ln_kernel<<<MROWS, 256, 0, stream>>>(x2, n2w, n2b, h2bf);
  gemm_bf16_kernel<<<dim3(MROWS / 128, 4 * CDIM / 128), 256, 0, stream>>>(
      h2bf, CDIM, fc1w_bf, fc1_b, nullptr, nullptr, fc1a_bf, MROWS, 4 * CDIM, CDIM, 1);
  gemm_bf16_kernel<<<dim3(MROWS / 128, CDIM / 128), 256, 0, stream>>>(
      fc1a_bf, 4 * CDIM, fc2w_bf, fc2_b, x2, out, nullptr, MROWS, CDIM, 4 * CDIM, 0);
}
